// GBKGNN_57097295233459
// MI455X (gfx1250) — matrix operand order repaired and checked by reading, compiled, not run on hardware
//
#include <hip/hip_runtime.h>

typedef __attribute__((ext_vector_type(16))) _Float16 v16h;
typedef __attribute__((ext_vector_type(8)))  _Float16 v8h;
typedef __attribute__((ext_vector_type(4)))  _Float16 v4h;
typedef __attribute__((ext_vector_type(8)))  float    v8f;

#define N_NODES 50000
#define N_EDGES 800000

static_assert(N_NODES % 16 == 0, "M must be a multiple of 16 (no GEMM tail)");

// ---------------------------------------------------------------------------
// fp32 -> fp16 conversion, float4 -> half4 vectorized (n must be multiple of 4)
// ---------------------------------------------------------------------------
__global__ void gbk_cvt_f16(const float* __restrict__ src,
                            _Float16* __restrict__ dst, int n4) {
    int i = blockIdx.x * blockDim.x + threadIdx.x;
    if (i >= n4) return;
    float4 v = ((const float4*)src)[i];
    v4h h = { (_Float16)v.x, (_Float16)v.y, (_Float16)v.z, (_Float16)v.w };
    ((v4h*)dst)[i] = h;
}

// ---------------------------------------------------------------------------
// Pack layer-1 weights (cols: [0,128)=W_l1, [128,256)=W_p1, [256,384)=W_r1,
// 384=W_g1 dst-half, 385=W_g1 src-half, rest zero) into per-lane WMMA
// B-fragment order: Bp[((c*NT + nt)*32 + lane)*16 + j] = B[k][n],
// k = c*32 + (lane>=16 ? 16 : 0) + j, n = nt*16 + (lane&15).
// ---------------------------------------------------------------------------
__global__ void gbk_pack_w1(const float* __restrict__ Wl, const float* __restrict__ Wp,
                            const float* __restrict__ Wr, const float* __restrict__ Wg,
                            _Float16* __restrict__ Bp) {
    const int NT = 25;                       // 400 columns
    int idx = blockIdx.x * blockDim.x + threadIdx.x;
    if (idx >= 4 * NT * 32 * 16) return;
    int j    = idx & 15;
    int lane = (idx >> 4) & 31;
    int t    = idx >> 9;
    int nt   = t % NT;
    int c    = t / NT;
    int k = c * 32 + ((lane >> 4) << 4) + j;   // 0..127
    int n = nt * 16 + (lane & 15);             // 0..399
    float v = 0.f;
    if      (n < 128) v = Wl[k * 128 + n];
    else if (n < 256) v = Wp[k * 128 + (n - 128)];
    else if (n < 384) v = Wr[k * 128 + (n - 256)];
    else if (n == 384) v = Wg[k];              // gate dst half
    else if (n == 385) v = Wg[128 + k];        // gate src half
    Bp[idx] = (_Float16)v;
}

// Layer-2 weights (cols: [0,16)=W_l2, [16,32)=W_p2, [32,48)=W_r2,
// 48=W_g2 dst-half, 49=W_g2 src-half, rest zero). 64 columns -> 4 n-tiles.
__global__ void gbk_pack_w2(const float* __restrict__ Wl, const float* __restrict__ Wp,
                            const float* __restrict__ Wr, const float* __restrict__ Wg,
                            _Float16* __restrict__ Bp) {
    const int NT = 4;
    int idx = blockIdx.x * blockDim.x + threadIdx.x;
    if (idx >= 4 * NT * 32 * 16) return;
    int j    = idx & 15;
    int lane = (idx >> 4) & 31;
    int t    = idx >> 9;
    int nt   = t % NT;
    int c    = t / NT;
    int k = c * 32 + ((lane >> 4) << 4) + j;
    int n = nt * 16 + (lane & 15);
    float v = 0.f;
    if      (n < 16) v = Wl[k * 16 + n];
    else if (n < 32) v = Wp[k * 16 + (n - 16)];
    else if (n < 48) v = Wr[k * 16 + (n - 32)];
    else if (n == 48) v = Wg[k];
    else if (n == 49) v = Wg[128 + k];
    Bp[idx] = (_Float16)v;
}

__global__ void gbk_bias1(const float* __restrict__ bl, const float* __restrict__ bp,
                          float* __restrict__ bias) {
    int n = blockIdx.x * blockDim.x + threadIdx.x;
    if (n >= 400) return;
    float v = 0.f;
    if (n < 128) v = bl[n];
    else if (n < 256) v = bp[n - 128];
    bias[n] = v;
}

__global__ void gbk_bias2(const float* __restrict__ bl, const float* __restrict__ bp,
                          float* __restrict__ bias) {
    int n = threadIdx.x;
    if (n >= 64) return;
    float v = 0.f;
    if (n < 16) v = bl[n];
    else if (n < 32) v = bp[n - 16];
    bias[n] = v;
}

// ---------------------------------------------------------------------------
// WMMA GEMM: C[M x (NTILES*16)] = A[M x 128](f16) * Bpack + bias.
// 8 waves per block; each wave owns one 16-row M-tile, preloads its 4 A
// fragments (K chunks of 32) and sweeps all N-tiles with v_wmma_f32_16x16x32_f16.
// NTILES is a template constant so the store epilogue uses immediate offsets
// (row stride N known at compile time) and has no bounds branches (M % 16 == 0).
// ---------------------------------------------------------------------------
template <int NTILES>
__global__ __launch_bounds__(256)
void gbk_wmma_gemm(const _Float16* __restrict__ A,
                   const _Float16* __restrict__ Bp,
                   const float* __restrict__ bias,
                   float* __restrict__ C, int M) {
    constexpr int N = NTILES * 16;
    const int lane  = threadIdx.x & 31;
    const int wave  = threadIdx.x >> 5;
    const int mtile = blockIdx.x * 8 + wave;
    if (mtile * 16 >= M) return;           // whole-wave uniform exit only
    const int row0 = mtile * 16;
    const int hi   = lane >> 4;            // half-wave selector
    const int r    = row0 + (lane & 15);

    // A fragments: lane<16 holds K c*32+0..15; lane>=16 holds K c*32+16..31.
    union { v16h v; v8h h[2]; } a[4];
    const _Float16* arow = A + (size_t)r * 128;
#pragma unroll
    for (int c = 0; c < 4; ++c) {
        const _Float16* p = arow + c * 32 + hi * 16;
        a[c].h[0] = *(const v8h*)(p);        // K = c*32 + hi*16 .. +7
        a[c].h[1] = *(const v8h*)(p + 8);    // K = c*32 + hi*16 + 8 .. +15
    }

    const int colLane = lane & 15;
    for (int nt = 0; nt < NTILES; ++nt) {
        v8f acc = {};
#pragma unroll
        for (int c = 0; c < 4; ++c) {
            v16h b = *(const v16h*)(Bp + (((size_t)c * NTILES + nt) * 32 + lane) * 16);
            acc = __builtin_amdgcn_wmma_f32_16x16x32_f16(
                false, a[c].v, false, b, (short)0, acc, false, false);
        }
        const float bv = bias[nt * 16 + colLane];
        // C frag: VGPR v -> row row0 + hi*8 + v, col nt*16 + colLane.
        float* cptr = C + (size_t)(row0 + hi * 8) * N + nt * 16 + colLane;
#pragma unroll
        for (int v = 0; v < 8; ++v)
            cptr[(size_t)v * N] = acc[v] + bv;   // immediate-offset stores
    }
}

// ---------------------------------------------------------------------------
// Layer-1 edge phase: one wave32 per edge, lane covers 4 of 128 dims (float4).
// H1 row layout (stride 400): [0,128)=h_l, [128,256)=h_p, [256,384)=x@W_r,
// 384=g_i(dst half), 385=g_j(src half).
// ---------------------------------------------------------------------------
__global__ __launch_bounds__(256)
void gbk_edge_l1(const long long* __restrict__ ei, const float* __restrict__ H,
                 const float* __restrict__ bg, float* __restrict__ agg,
                 float* __restrict__ deg, int E) {
    const int e = blockIdx.x * 8 + (threadIdx.x >> 5);
    if (e >= E) return;
    const int lane = threadIdx.x & 31;
    const long long s = ei[e];                 // source (j)
    const long long d = ei[(size_t)E + e];     // destination (i)
    const float* hs = H + (size_t)s * 400;
    const float z = H[(size_t)d * 400 + 384] + hs[385] + bg[0];
    const float sig = 1.f / (1.f + __expf(-z));
    const float4 hl = *(const float4*)(hs + lane * 4);
    const float4 hp = *(const float4*)(hs + 128 + lane * 4);
    float* ap = agg + (size_t)d * 128 + lane * 4;
    atomicAdd(ap + 0, sig * hl.x + (1.f - sig) * hp.x);
    atomicAdd(ap + 1, sig * hl.y + (1.f - sig) * hp.y);
    atomicAdd(ap + 2, sig * hl.z + (1.f - sig) * hp.z);
    atomicAdd(ap + 3, sig * hl.w + (1.f - sig) * hp.w);
    if (lane == 0) atomicAdd(deg + d, 1.0f);
}

// mean + residual + ReLU, emit f16 input for the layer-2 GEMM; 4 dims/thread
__global__ void gbk_fin_l1(const float* __restrict__ agg, const float* __restrict__ deg,
                           const float* __restrict__ H, _Float16* __restrict__ hh,
                           int total4) {
    int i = blockIdx.x * blockDim.x + threadIdx.x;
    if (i >= total4) return;
    int nd = i >> 5;           // (i*4)/128
    int k  = (i & 31) * 4;
    float dg = deg[nd]; if (dg < 1.f) dg = 1.f;
    float4 ag = ((const float4*)agg)[i];
    const float* hr = H + (size_t)nd * 400 + 256 + k;
    float4 rv = *(const float4*)hr;
    float vx = ag.x / dg + rv.x, vy = ag.y / dg + rv.y;
    float vz = ag.z / dg + rv.z, vw = ag.w / dg + rv.w;
    v4h h = { (_Float16)(vx > 0.f ? vx : 0.f), (_Float16)(vy > 0.f ? vy : 0.f),
              (_Float16)(vz > 0.f ? vz : 0.f), (_Float16)(vw > 0.f ? vw : 0.f) };
    ((v4h*)hh)[i] = h;
}

// Layer-2 edge phase: 16 lanes per edge (dim = lane&15).
// H2 row layout (stride 64): [0,16)=h_l, [16,32)=h_p, [32,48)=h@W_r, 48=g_i, 49=g_j.
__global__ __launch_bounds__(256)
void gbk_edge_l2(const long long* __restrict__ ei, const float* __restrict__ H2,
                 const float* __restrict__ bg, float* __restrict__ agg2, int E) {
    int t = blockIdx.x * 256 + threadIdx.x;
    int e = t >> 4;
    if (e >= E) return;
    int k = t & 15;
    const long long s = ei[e];
    const long long d = ei[(size_t)E + e];
    const float* hs = H2 + (size_t)s * 64;
    const float z = H2[(size_t)d * 64 + 48] + hs[49] + bg[0];
    const float sig = 1.f / (1.f + __expf(-z));
    const float hl = hs[k];
    const float hp = hs[16 + k];
    atomicAdd(agg2 + (size_t)d * 16 + k, sig * hl + (1.f - sig) * hp);
}

// mean + residual, final fp32 output; 4 dims/thread
__global__ void gbk_fin_l2(const float* __restrict__ agg2, const float* __restrict__ deg,
                           const float* __restrict__ H2, float* __restrict__ out,
                           int total4) {
    int i = blockIdx.x * blockDim.x + threadIdx.x;
    if (i >= total4) return;
    int nd = i >> 2;           // (i*4)/16
    int k  = (i & 3) * 4;
    float dg = deg[nd]; if (dg < 1.f) dg = 1.f;
    float4 ag = ((const float4*)agg2)[i];
    float4 rv = *(const float4*)(H2 + (size_t)nd * 64 + 32 + k);
    float4 o = { ag.x / dg + rv.x, ag.y / dg + rv.y,
                 ag.z / dg + rv.z, ag.w / dg + rv.w };
    ((float4*)out)[i] = o;
}

// ---------------------------------------------------------------------------
extern "C" void kernel_launch(void* const* d_in, const int* in_sizes, int n_in,
                              void* d_out, int out_size, void* d_ws, size_t ws_size,
                              hipStream_t stream) {
    (void)in_sizes; (void)n_in; (void)out_size; (void)ws_size;

    const float*     x    = (const float*)d_in[0];
    const long long* ei   = (const long long*)d_in[1];   // int64 edge_index [2, E]
    const float* W_l1 = (const float*)d_in[2];
    const float* b_l1 = (const float*)d_in[3];
    const float* W_p1 = (const float*)d_in[4];
    const float* b_p1 = (const float*)d_in[5];
    const float* W_r1 = (const float*)d_in[6];
    const float* W_g1 = (const float*)d_in[7];
    const float* b_g1 = (const float*)d_in[8];
    const float* W_l2 = (const float*)d_in[9];
    const float* b_l2 = (const float*)d_in[10];
    const float* W_p2 = (const float*)d_in[11];
    const float* b_p2 = (const float*)d_in[12];
    const float* W_r2 = (const float*)d_in[13];
    const float* W_g2 = (const float*)d_in[14];
    const float* b_g2 = (const float*)d_in[15];

    char* ws = (char*)d_ws;
    size_t off = 0;
    auto alloc = [&](size_t bytes) -> char* {
        char* p = ws + off;
        off = (off + bytes + 255) & ~(size_t)255;
        return p;
    };

    _Float16* xh    = (_Float16*)alloc((size_t)N_NODES * 128 * 2);
    _Float16* Bp1   = (_Float16*)alloc((size_t)4 * 25 * 32 * 16 * 2);
    float*    bias1 = (float*)   alloc(400 * 4);
    float*    H1    = (float*)   alloc((size_t)N_NODES * 400 * 4);
    float*    agg1  = (float*)   alloc((size_t)N_NODES * 128 * 4);
    float*    deg   = (float*)   alloc((size_t)N_NODES * 4);
    _Float16* hh    = (_Float16*)alloc((size_t)N_NODES * 128 * 2);
    _Float16* Bp2   = (_Float16*)alloc((size_t)4 * 4 * 32 * 16 * 2);
    float*    bias2 = (float*)   alloc(64 * 4);
    float*    H2    = (float*)   alloc((size_t)N_NODES * 64 * 4);
    float*    agg2  = (float*)   alloc((size_t)N_NODES * 16 * 4);

    hipMemsetAsync(agg1, 0, (size_t)N_NODES * 128 * 4, stream);
    hipMemsetAsync(deg,  0, (size_t)N_NODES * 4, stream);
    hipMemsetAsync(agg2, 0, (size_t)N_NODES * 16 * 4, stream);

    // ---- Layer 1 ----
    gbk_cvt_f16<<<(N_NODES * 128 / 4 + 255) / 256, 256, 0, stream>>>(x, xh, N_NODES * 128 / 4);
    gbk_pack_w1<<<(4 * 25 * 32 * 16 + 255) / 256, 256, 0, stream>>>(W_l1, W_p1, W_r1, W_g1, Bp1);
    gbk_bias1<<<2, 256, 0, stream>>>(b_l1, b_p1, bias1);

    const int mtiles = N_NODES / 16;               // 3125
    gbk_wmma_gemm<25><<<(mtiles + 7) / 8, 256, 0, stream>>>(xh, Bp1, bias1, H1, N_NODES);

    gbk_edge_l1<<<(N_EDGES + 7) / 8, 256, 0, stream>>>(ei, H1, b_g1, agg1, deg, N_EDGES);
    gbk_fin_l1<<<(N_NODES * 128 / 4 + 255) / 256, 256, 0, stream>>>(agg1, deg, H1, hh,
                                                                    N_NODES * 128 / 4);

    // ---- Layer 2 ----
    gbk_pack_w2<<<(4 * 4 * 32 * 16 + 255) / 256, 256, 0, stream>>>(W_l2, W_p2, W_r2, W_g2, Bp2);
    gbk_bias2<<<1, 64, 0, stream>>>(b_l2, b_p2, bias2);

    gbk_wmma_gemm<4><<<(mtiles + 7) / 8, 256, 0, stream>>>(hh, Bp2, bias2, H2, N_NODES);

    gbk_edge_l2<<<((size_t)N_EDGES * 16 + 255) / 256, 256, 0, stream>>>(ei, H2, b_g2, agg2, N_EDGES);
    gbk_fin_l2<<<(N_NODES * 16 / 4 + 255) / 256, 256, 0, stream>>>(agg2, deg, H2, (float*)d_out,
                                                                   N_NODES * 16 / 4);
}
